// ScaledDotProductAttention_14001593385058
// MI455X (gfx1250) — matrix operand order repaired and checked by reading, not compiled, not run on hardware
//
#include <hip/hip_runtime.h>

typedef __attribute__((ext_vector_type(16))) _Float16 v16h;
typedef __attribute__((ext_vector_type(8)))  _Float16 v8h;
typedef __attribute__((ext_vector_type(4)))  _Float16 v4h;
typedef __attribute__((ext_vector_type(8)))  float    v8f;

#define CAT16(a,b) __builtin_shufflevector((a),(b),0,1,2,3,4,5,6,7,8,9,10,11,12,13,14,15)

static constexpr int BS = 8, QL = 2048, KL = 2048, D = 512;
static constexpr int BQ = 16;            // query rows per block
static constexpr int BK = 256;           // keys per outer iteration (8 waves * 32)
static constexpr float NEG_INF_C = -1e9f;
static constexpr int SSTR = 260;         // padded S row stride (floats)  -> conflict-free row reads
static constexpr int PSTR = 264;         // padded P row stride (halves)  -> conflict-free frag reads

// ---------- prep kernel: K fp32 -> fp16 (same layout) ----------
__global__ void kconv(const float* __restrict__ X, _Float16* __restrict__ Y) {
  size_t i4 = (size_t)blockIdx.x * blockDim.x + threadIdx.x;
  float4 v = ((const float4*)X)[i4];
  v4h h = { (_Float16)v.x, (_Float16)v.y, (_Float16)v.z, (_Float16)v.w };
  ((v4h*)Y)[i4] = h;
}

// ---------- prep kernel: V fp32 [b][k][d] -> fp16 Vt [b][d][k] ----------
__global__ void vtrans(const float* __restrict__ V, _Float16* __restrict__ Vt) {
  __shared__ float tile[32][33];
  const int b  = blockIdx.z;
  const int k0 = blockIdx.x * 32;
  const int d0 = blockIdx.y * 32;
  const int c  = threadIdx.x & 31;
  const int r0 = threadIdx.x >> 5;
#pragma unroll
  for (int rr = r0; rr < 32; rr += 8)
    tile[rr][c] = V[((size_t)b * KL + k0 + rr) * D + d0 + c];
  __syncthreads();
#pragma unroll
  for (int rr = r0; rr < 32; rr += 8)
    Vt[((size_t)b * D + d0 + rr) * KL + k0 + c] = (_Float16)tile[c][rr];
}

// ---------- flash attention forward ----------
__global__ __launch_bounds__(256, 1) void fa_fwd(
    const float* __restrict__ Q, const _Float16* __restrict__ Kh,
    const _Float16* __restrict__ Vt, float* __restrict__ O,
    const int* __restrict__ umP, const int* __restrict__ cdP)
{
  __shared__ float    Sb[BQ][SSTR];   // raw scores (per-wave private 32-col chunks)
  __shared__ _Float16 Pb[BQ][PSTR];   // softmax numerators (f16)
  __shared__ float    Lmx[BQ][20];    // per-chunk row maxima
  __shared__ float    Lsm[BQ][20];    // per-chunk row partial sums
  __shared__ float    Carr[16];       // per-row rescale factor
  __shared__ float    Larr[16];       // final row sums

  const int tid  = threadIdx.x;
  const int wave = tid >> 5;
  const int lane = tid & 31;
  const int r16  = lane & 15;
  const int hi   = lane >> 4;
  const int off8 = hi * 8;
  const int kbA  = hi * 8;            // A-fragment K base
  const int q0   = blockIdx.x * BQ;
  const int b    = blockIdx.y;
  const int use_mask = *umP;
  const float scale  = rsqrtf((float)(*cdP));

  // ---- load Q fragments once (A-matrix layout, f32 -> f16) ----
  const float* Qrow = Q + ((size_t)b * QL + q0 + r16) * D;
  v16h qf[16];
#pragma unroll
  for (int ks = 0; ks < 16; ++ks) {
    const int d0 = ks * 32;
    const float4* p0 = (const float4*)(Qrow + d0 + kbA);
    const float4* p1 = (const float4*)(Qrow + d0 + kbA + 16);
    float4 a0 = p0[0], a1 = p0[1], c0 = p1[0], c1 = p1[1];
    v16h f;
    f[0]=(_Float16)a0.x;  f[1]=(_Float16)a0.y;  f[2]=(_Float16)a0.z;  f[3]=(_Float16)a0.w;
    f[4]=(_Float16)a1.x;  f[5]=(_Float16)a1.y;  f[6]=(_Float16)a1.z;  f[7]=(_Float16)a1.w;
    f[8]=(_Float16)c0.x;  f[9]=(_Float16)c0.y;  f[10]=(_Float16)c0.z; f[11]=(_Float16)c0.w;
    f[12]=(_Float16)c1.x; f[13]=(_Float16)c1.y; f[14]=(_Float16)c1.z; f[15]=(_Float16)c1.w;
    qf[ks] = f;
  }

  v8f acc[4] = {};
  float m_row = -3.0e38f;
  float l_row = 0.0f;

  const int iters = use_mask ? ((q0 + BQ - 1) / BK + 1) : (KL / BK);

  for (int it = 0; it < iters; ++it) {
    const int kt0 = it * BK;
    const int keyBase = kt0 + wave * 32;

    // ---- Phase A: S chunk = Q * K^T (wave-private key chunk, no redundancy) ----
    v8f sacc[2] = {};
#pragma unroll
    for (int nt = 0; nt < 2; ++nt) {
      const _Float16* kp = Kh + ((size_t)b * KL + keyBase + nt * 16 + r16) * D + hi * 8;
#pragma unroll
      for (int ks = 0; ks < 16; ++ks) {
        const v8h* bp = (const v8h*)(kp + ks * 32);
        v16h bf = CAT16(bp[0], bp[2]);
        sacc[nt] = __builtin_amdgcn_wmma_f32_16x16x32_f16(
            false, qf[ks], false, bf, (short)0, sacc[nt], false, false);
      }
    }
#pragma unroll
    for (int nt = 0; nt < 2; ++nt)
#pragma unroll
      for (int i = 0; i < 8; ++i)
        Sb[i + off8][wave * 32 + nt * 16 + r16] = sacc[nt][i];

    // ---- B1: masked, scaled logits + local row max (same-wave LDS transpose) ----
    float lg[16];
    {
      const float4* sp = (const float4*)&Sb[r16][wave * 32 + hi * 16];
      float4 s0 = sp[0], s1 = sp[1], s2 = sp[2], s3 = sp[3];
      lg[0]=s0.x;  lg[1]=s0.y;  lg[2]=s0.z;  lg[3]=s0.w;
      lg[4]=s1.x;  lg[5]=s1.y;  lg[6]=s1.z;  lg[7]=s1.w;
      lg[8]=s2.x;  lg[9]=s2.y;  lg[10]=s2.z; lg[11]=s2.w;
      lg[12]=s3.x; lg[13]=s3.y; lg[14]=s3.z; lg[15]=s3.w;
    }
    const int colBase = kt0 + wave * 32 + hi * 16;
    const int qrow = q0 + r16;
    float lm = -3.0e38f;
#pragma unroll
    for (int j = 0; j < 16; ++j) {
      float s = lg[j];
      if (use_mask && (colBase + j > qrow)) s += NEG_INF_C;   // matches reference: (mm + mask*NEG_INF)
      s *= scale;
      lg[j] = s;
      lm = fmaxf(lm, s);
    }
    Lmx[r16][wave * 2 + hi] = lm;
    __syncthreads();

    // ---- B2: global row max, exp, write P (f16), partial sums ----
    float tmax;
    {
      const float4* mp = (const float4*)&Lmx[r16][0];
      float4 m0 = mp[0], m1 = mp[1], m2 = mp[2], m3 = mp[3];
      tmax = fmaxf(fmaxf(fmaxf(m0.x, m0.y), fmaxf(m0.z, m0.w)),
             fmaxf(fmaxf(fmaxf(m1.x, m1.y), fmaxf(m1.z, m1.w)),
             fmaxf(fmaxf(fmaxf(m2.x, m2.y), fmaxf(m2.z, m2.w)),
                   fmaxf(fmaxf(m3.x, m3.y), fmaxf(m3.z, m3.w)))));
    }
    const float m_new = fmaxf(m_row, tmax);
    const float crow  = __expf(m_row - m_new);
    m_row = m_new;
    float ls = 0.0f;
    _Float16 ph[16];
#pragma unroll
    for (int j = 0; j < 16; ++j) {
      float p = __expf(lg[j] - m_new);
      ls += p;
      ph[j] = (_Float16)p;
    }
    {
      v8h* pp = (v8h*)&Pb[r16][wave * 32 + hi * 16];
      v8h p0 = { ph[0], ph[1], ph[2],  ph[3],  ph[4],  ph[5],  ph[6],  ph[7]  };
      v8h p1 = { ph[8], ph[9], ph[10], ph[11], ph[12], ph[13], ph[14], ph[15] };
      pp[0] = p0; pp[1] = p1;
    }
    Lsm[r16][wave * 2 + hi] = ls;
    if (tid < 16) Carr[tid] = crow;
    __syncthreads();

    // ---- B3: rescale accumulators; wave0 lanes update l ----
    {
      const float4* cp = (const float4*)&Carr[off8];
      float4 c0 = cp[0], c1 = cp[1];
      float cv[8] = { c0.x, c0.y, c0.z, c0.w, c1.x, c1.y, c1.z, c1.w };
#pragma unroll
      for (int t = 0; t < 4; ++t)
#pragma unroll
        for (int i = 0; i < 8; ++i)
          acc[t][i] *= cv[i];
    }
    if (tid < 16) {
      const float4* sp = (const float4*)&Lsm[r16][0];
      float4 s0 = sp[0], s1 = sp[1], s2 = sp[2], s3 = sp[3];
      float ts = (s0.x + s0.y + s0.z + s0.w) + (s1.x + s1.y + s1.z + s1.w)
               + (s2.x + s2.y + s2.z + s2.w) + (s3.x + s3.y + s3.z + s3.w);
      l_row = l_row * crow + ts;
    }

    // ---- PV: O_slice += P(16x256) * V(256x64)  (wave-private d-slice) ----
#pragma unroll
    for (int kc = 0; kc < 8; ++kc) {
      const v8h* ap0 = (const v8h*)&Pb[r16][kc * 32 + kbA];
      const v8h* ap1 = (const v8h*)&Pb[r16][kc * 32 + 16 + kbA];
      v16h af = CAT16(ap0[0], ap1[0]);
#pragma unroll
      for (int t = 0; t < 4; ++t) {
        const _Float16* vp = Vt + ((size_t)b * D + wave * 64 + t * 16 + r16) * KL
                                + kt0 + kc * 32 + hi * 8;
        const v8h* bp = (const v8h*)vp;
        v16h bf = CAT16(bp[0], bp[2]);
        acc[t] = __builtin_amdgcn_wmma_f32_16x16x32_f16(
            false, af, false, bf, (short)0, acc[t], false, false);
      }
    }
  }

  // ---- final normalization and store ----
  if (tid < 16) Larr[tid] = l_row;
  __syncthreads();
  float rinv[8];
  {
    const float4* lp = (const float4*)&Larr[off8];
    float4 l0 = lp[0], l1 = lp[1];
    float lv[8] = { l0.x, l0.y, l0.z, l0.w, l1.x, l1.y, l1.z, l1.w };
#pragma unroll
    for (int i = 0; i < 8; ++i) rinv[i] = 1.0f / lv[i];
  }
#pragma unroll
  for (int t = 0; t < 4; ++t)
#pragma unroll
    for (int i = 0; i < 8; ++i)
      O[((size_t)b * QL + q0 + i + off8) * D + wave * 64 + t * 16 + r16] = acc[t][i] * rinv[i];
}

extern "C" void kernel_launch(void* const* d_in, const int* in_sizes, int n_in,
                              void* d_out, int out_size, void* d_ws, size_t ws_size,
                              hipStream_t stream) {
  (void)in_sizes; (void)n_in; (void)out_size; (void)ws_size;
  const float* Q = (const float*)d_in[0];
  const float* K = (const float*)d_in[1];
  const float* V = (const float*)d_in[2];
  const int*  um = (const int*)d_in[3];
  const int*  cd = (const int*)d_in[4];

  _Float16* Kh = (_Float16*)d_ws;                       // [BS][KL][D] f16
  _Float16* Vt = Kh + (size_t)BS * KL * D;              // [BS][D][KL] f16

  const int nK4 = BS * KL * D / 4;
  kconv<<<nK4 / 256, 256, 0, stream>>>(K, Kh);
  vtrans<<<dim3(KL / 32, D / 32, BS), 256, 0, stream>>>(V, Vt);
  fa_fwd<<<dim3(QL / BQ, BS), 256, 0, stream>>>(Q, Kh, Vt, (float*)d_out, um, cd);
}
